// MultiHeadSelfAttention_1211180777687
// MI455X (gfx1250) — compile-verified, matrix operand order checked by reading, not run on hardware
//
#include <hip/hip_runtime.h>

#define SEQ 4096
#define DIM 768
#define NQKV 2304
#define HEADS 12
#define HD 64
#define QK_SCALE 0.125f   // 64^-0.5

typedef __attribute__((ext_vector_type(16))) _Float16 v16h;
typedef __attribute__((ext_vector_type(8)))  _Float16 v8h;
typedef __attribute__((ext_vector_type(8)))  float    v8f;

// ---------------------------------------------------------------------------
// CDNA5 async global->LDS copy (ASYNCcnt path), with safe fallback.
// ---------------------------------------------------------------------------
#if defined(__has_builtin)
#if __has_builtin(__builtin_amdgcn_global_load_async_to_lds_b128) && \
    __has_builtin(__builtin_amdgcn_s_wait_asynccnt)
#define USE_ASYNC 1
#endif
#endif
#ifndef USE_ASYNC
#define USE_ASYNC 0
#endif

typedef int v4i __attribute__((vector_size(4 * sizeof(int))));

#if USE_ASYNC
typedef __attribute__((address_space(1))) v4i* g_v4i_ptr;
typedef __attribute__((address_space(3))) v4i* l_v4i_ptr;
__device__ __forceinline__ void cp_b128(const void* g, void* l) {
  __builtin_amdgcn_global_load_async_to_lds_b128((g_v4i_ptr)g, (l_v4i_ptr)l, 0, 0);
}
__device__ __forceinline__ void cp_wait() { __builtin_amdgcn_s_wait_asynccnt(0); }
#else
__device__ __forceinline__ void cp_b128(const void* g, void* l) {
  *(uint4*)l = *(const uint4*)g;
}
__device__ __forceinline__ void cp_wait() {}
#endif

// ---------------------------------------------------------------------------
// WMMA helpers
// ---------------------------------------------------------------------------
__device__ __forceinline__ v8f wmma_f16(v16h a, v16h b, v8f c) {
  return __builtin_amdgcn_wmma_f32_16x16x32_f16(false, a, false, b,
                                                (short)0, c, false, false);
}

// 16x32 (MxK) f16 fragment from LDS [row][k] with stride ld (CDNA5 7.12.2):
//   lanes 0-15 : M = lane,    elems 0-7 -> K=0..7,   8-15 -> K=16..23
//   lanes 16-31: M = lane-16, elems 0-7 -> K=8..15,  8-15 -> K=24..31
// B fragments mirror with N in place of M, so B tiles are stored [n][k].
__device__ __forceinline__ v16h load_frag16(const _Float16* base, int ld,
                                            int rowbase, int kofs) {
  const int lane = threadIdx.x & 31;
  const int r  = rowbase + (lane & 15);
  const int kb = kofs + ((lane >> 4) << 3);
  const _Float16* p = base + r * ld + kb;
  v8h lo = *(const v8h*)(p);
  v8h hi = *(const v8h*)(p + 16);
  v16h out;
#pragma unroll
  for (int e = 0; e < 8; ++e) { out[e] = lo[e]; out[e + 8] = hi[e]; }
  return out;
}

// ---------------------------------------------------------------------------
// Kernel 1: QKV = x @ w_qkv + b_qkv -> per-head f16 Q/K/V (Q pre-scaled).
// 256 thr / 8 waves. Block tile 128x64, K-step 32, register double-buffered.
// ---------------------------------------------------------------------------
__global__ __launch_bounds__(256) void qkv_gemm_kernel(
    const float* __restrict__ x, const float* __restrict__ w,
    const float* __restrict__ bias, _Float16* __restrict__ qbuf,
    _Float16* __restrict__ kbuf, _Float16* __restrict__ vbuf) {
  __shared__ _Float16 As[128 * 40];   // [m][k], stride 40 halves (80B)
  __shared__ _Float16 Bs[64 * 40];    // transposed [n][k]

  const int tid  = threadIdx.x;
  const int lane = tid & 31;
  const int wid  = tid >> 5;
  const int n0 = blockIdx.x * 64;
  const int m0 = blockIdx.y * 128;
  const int wr = wid >> 1;   // 0..3
  const int wc = wid & 1;    // 0..1

  float4 ra[4];   // A: 128x32 f32 = 1024 float4 -> 4/thread
  float4 rb[2];   // B: 32x64  f32 = 512 float4  -> 2/thread

  auto load_regs = [&](int k0) {
#pragma unroll
    for (int j = 0; j < 4; ++j) {
      int v = tid + j * 256;
      int r = v >> 3, c = (v & 7) * 4;
      ra[j] = *(const float4*)&x[(size_t)(m0 + r) * DIM + k0 + c];
    }
#pragma unroll
    for (int j = 0; j < 2; ++j) {
      int v = tid + j * 256;
      int k = v >> 4, n = (v & 15) * 4;
      rb[j] = *(const float4*)&w[(size_t)(k0 + k) * NQKV + n0 + n];
    }
  };
  auto store_lds = [&]() {
#pragma unroll
    for (int j = 0; j < 4; ++j) {
      int v = tid + j * 256;
      int r = v >> 3, c = (v & 7) * 4;
      _Float16* p = &As[r * 40 + c];
      p[0] = (_Float16)ra[j].x; p[1] = (_Float16)ra[j].y;
      p[2] = (_Float16)ra[j].z; p[3] = (_Float16)ra[j].w;
    }
#pragma unroll
    for (int j = 0; j < 2; ++j) {
      int v = tid + j * 256;
      int k = v >> 4, n = (v & 15) * 4;
      Bs[(n + 0) * 40 + k] = (_Float16)rb[j].x;
      Bs[(n + 1) * 40 + k] = (_Float16)rb[j].y;
      Bs[(n + 2) * 40 + k] = (_Float16)rb[j].z;
      Bs[(n + 3) * 40 + k] = (_Float16)rb[j].w;
    }
  };

  v8f acc[4] = {v8f{}, v8f{}, v8f{}, v8f{}};

  load_regs(0);
  for (int k0 = 0; k0 < DIM; k0 += 32) {
    store_lds();
    __syncthreads();
    if (k0 + 32 < DIM) load_regs(k0 + 32);   // overlap with WMMA below
    v16h a0 = load_frag16(As, 40, wr * 32, 0);
    v16h a1 = load_frag16(As, 40, wr * 32 + 16, 0);
    v16h b0 = load_frag16(Bs, 40, wc * 32, 0);
    v16h b1 = load_frag16(Bs, 40, wc * 32 + 16, 0);
    acc[0] = wmma_f16(a0, b0, acc[0]);
    acc[1] = wmma_f16(a0, b1, acc[1]);
    acc[2] = wmma_f16(a1, b0, acc[2]);
    acc[3] = wmma_f16(a1, b1, acc[3]);
    __syncthreads();
  }

  const int nlo  = lane & 15;
  const int mofs = (lane >> 4) << 3;
#pragma unroll
  for (int ar = 0; ar < 2; ++ar) {
#pragma unroll
    for (int bc = 0; bc < 2; ++bc) {
      v8f a = acc[ar * 2 + bc];
#pragma unroll
      for (int i = 0; i < 8; ++i) {
        int m = m0 + wr * 32 + ar * 16 + mofs + i;
        int n = n0 + wc * 32 + bc * 16 + nlo;
        float vv = a[i] + bias[n];
        int sel    = n / DIM;
        int within = n - sel * DIM;
        int head   = within >> 6;
        int hd     = within & 63;
        if (sel == 0) vv *= QK_SCALE;
        _Float16* dst = (sel == 0) ? qbuf : (sel == 1) ? kbuf : vbuf;
        dst[(head * SEQ + m) * HD + hd] = (_Float16)vv;
      }
    }
  }
}

// ---------------------------------------------------------------------------
// Kernel 2: flash attention; block = (head, 64-query tile).
// Ping-pong K tiles via async global->LDS, V prefetched through registers:
// tile i+1 is in flight across the whole compute phase of tile i.
// ---------------------------------------------------------------------------
__global__ __launch_bounds__(256) void attn_kernel(
    const _Float16* __restrict__ qbuf, const _Float16* __restrict__ kbuf,
    const _Float16* __restrict__ vbuf, _Float16* __restrict__ obuf) {
  __shared__ _Float16 Qs[64 * 72];        // [query][hd]
  __shared__ _Float16 Ks[2][64 * 72];     // ping-pong [key][hd]
  __shared__ _Float16 Vt[64 * 72];        // [hd][key]
  __shared__ _Float16 Pt[64 * 72];        // f16 probabilities
  __shared__ float    Ss[64 * 68];        // fp32 scores
  __shared__ float mrow[64];
  __shared__ float lrow[64];
  __shared__ float frow[64];

  const int tid  = threadIdx.x;
  const int lane = tid & 31;
  const int wid  = tid >> 5;
  const int h  = blockIdx.y;
  const int q0 = blockIdx.x * 64;
  const int wr = wid >> 1;
  const int wc = wid & 1;
  const size_t hbase = (size_t)h * SEQ * HD;

  const int cr  = tid >> 3;         // chunk row (0..31), j adds +32
  const int cc8 = (tid & 7) * 8;    // chunk col in halves

  // Q tile: async once
#pragma unroll
  for (int j = 0; j < 2; ++j) {
    int r = cr + j * 32;
    cp_b128(&qbuf[hbase + (size_t)(q0 + r) * HD + cc8], &Qs[r * 72 + cc8]);
  }
  if (tid < 64) { mrow[tid] = -1e30f; lrow[tid] = 0.f; }

  uint4 rv[2];
  auto prefetch = [&](int kb0, int buf) {
#pragma unroll
    for (int j = 0; j < 2; ++j) {
      int r = cr + j * 32;
      cp_b128(&kbuf[hbase + (size_t)(kb0 + r) * HD + cc8], &Ks[buf][r * 72 + cc8]);
      rv[j] = *(const uint4*)&vbuf[hbase + (size_t)(kb0 + r) * HD + cc8];
    }
  };

  prefetch(0, 0);

  v8f o0 = {}; v8f o1 = {};
  const int nlo  = lane & 15;
  const int mofs = (lane >> 4) << 3;

  int cur = 0;
  for (int kb0 = 0; kb0 < SEQ; kb0 += 64, cur ^= 1) {
    cp_wait();                 // K tile (and Q on iter 0) resident
    // scatter prefetched V regs into Vt (prev PV consumers drained by loop-end barrier)
#pragma unroll
    for (int j = 0; j < 2; ++j) {
      int r = cr + j * 32;
      const _Float16* hv = (const _Float16*)&rv[j];
#pragma unroll
      for (int e = 0; e < 8; ++e) Vt[(cc8 + e) * 72 + r] = hv[e];
    }
    __syncthreads();           // publish Ks[cur], Vt

    if (kb0 + 64 < SEQ) prefetch(kb0 + 64, cur ^ 1);   // overlap with compute

    // S = Q K^T (Q pre-scaled)
    v8f s0 = {}; v8f s1 = {};
#pragma unroll
    for (int kk = 0; kk < HD; kk += 32) {
      v16h a  = load_frag16(Qs, 72, wr * 16, kk);
      v16h b0 = load_frag16(Ks[cur], 72, wc * 32, kk);
      v16h b1 = load_frag16(Ks[cur], 72, wc * 32 + 16, kk);
      s0 = wmma_f16(a, b0, s0);
      s1 = wmma_f16(a, b1, s1);
    }
#pragma unroll
    for (int i = 0; i < 8; ++i) {
      int r = wr * 16 + mofs + i;
      Ss[r * 68 + wc * 32 + nlo]      = s0[i];
      Ss[r * 68 + wc * 32 + 16 + nlo] = s1[i];
    }
    __syncthreads();

    // online softmax: one quad (same wave) per row
    {
      const int row  = tid >> 2;
      const int part = tid & 3;
      const float* srow = &Ss[row * 68 + part * 16];
      float mx = -1e30f;
#pragma unroll
      for (int e = 0; e < 16; ++e) mx = fmaxf(mx, srow[e]);
      mx = fmaxf(mx, __shfl_xor(mx, 1, 32));
      mx = fmaxf(mx, __shfl_xor(mx, 2, 32));
      float mprev = mrow[row];
      float mnew  = fmaxf(mprev, mx);
      float sum = 0.f;
      _Float16* prow = &Pt[row * 72 + part * 16];
#pragma unroll
      for (int e = 0; e < 16; ++e) {
        float p = __expf(srow[e] - mnew);
        sum += p;
        prow[e] = (_Float16)p;
      }
      sum += __shfl_xor(sum, 1, 32);
      sum += __shfl_xor(sum, 2, 32);
      if (part == 0) {
        float f = __expf(mprev - mnew);
        frow[row] = f;
        lrow[row] = lrow[row] * f + sum;
        mrow[row] = mnew;
      }
    }
    __syncthreads();

    // rescale running O, then O += P @ V
#pragma unroll
    for (int i = 0; i < 8; ++i) {
      float f = frow[wr * 16 + mofs + i];
      o0[i] *= f;
      o1[i] *= f;
    }
#pragma unroll
    for (int kk = 0; kk < 64; kk += 32) {
      v16h a  = load_frag16(Pt, 72, wr * 16, kk);
      v16h b0 = load_frag16(Vt, 72, wc * 32, kk);
      v16h b1 = load_frag16(Vt, 72, wc * 32 + 16, kk);
      o0 = wmma_f16(a, b0, o0);
      o1 = wmma_f16(a, b1, o1);
    }
    __syncthreads();           // drain Vt/Pt readers before next overwrite
  }

  // normalize, emit f16 [seq][dim]
#pragma unroll
  for (int i = 0; i < 8; ++i) {
    int r = wr * 16 + mofs + i;
    float inv = 1.f / lrow[r];
    size_t grow = (size_t)(q0 + r) * DIM + h * HD;
    obuf[grow + wc * 32 + nlo]      = (_Float16)(o0[i] * inv);
    obuf[grow + wc * 32 + 16 + nlo] = (_Float16)(o1[i] * inv);
  }
}

// ---------------------------------------------------------------------------
// Kernel 3: out = O @ w_proj + b_proj (fp32 out). Tile 128x64, K-step 32.
// Ping-pong LDS: A via async b128, B via register convert; next K-step's
// loads overlap the current WMMAs.
// ---------------------------------------------------------------------------
__global__ __launch_bounds__(256) void proj_gemm_kernel(
    const _Float16* __restrict__ o, const float* __restrict__ w,
    const float* __restrict__ bias, float* __restrict__ out) {
  __shared__ _Float16 As[2][128 * 40];
  __shared__ _Float16 Bs[2][64 * 40];   // transposed [n][k]

  const int tid  = threadIdx.x;
  const int lane = tid & 31;
  const int wid  = tid >> 5;
  const int n0 = blockIdx.x * 64;
  const int m0 = blockIdx.y * 128;
  const int wr = wid >> 1;
  const int wc = wid & 1;

  float4 rbv[2];
  auto issueA = [&](int k0, int buf) {
#pragma unroll
    for (int j = 0; j < 2; ++j) {
      int v = tid + j * 256;
      int r = v >> 2, c8 = (v & 3) * 8;
      cp_b128(&o[(size_t)(m0 + r) * DIM + k0 + c8], &As[buf][r * 40 + c8]);
    }
  };
  auto loadB = [&](int k0) {
#pragma unroll
    for (int j = 0; j < 2; ++j) {
      int v = tid + j * 256;
      int k = v >> 4, n = (v & 15) * 4;
      rbv[j] = *(const float4*)&w[(size_t)(k0 + k) * DIM + n0 + n];
    }
  };
  auto storeB = [&](int buf) {
#pragma unroll
    for (int j = 0; j < 2; ++j) {
      int v = tid + j * 256;
      int k = v >> 4, n = (v & 15) * 4;
      Bs[buf][(n + 0) * 40 + k] = (_Float16)rbv[j].x;
      Bs[buf][(n + 1) * 40 + k] = (_Float16)rbv[j].y;
      Bs[buf][(n + 2) * 40 + k] = (_Float16)rbv[j].z;
      Bs[buf][(n + 3) * 40 + k] = (_Float16)rbv[j].w;
    }
  };

  v8f acc[4] = {v8f{}, v8f{}, v8f{}, v8f{}};

  issueA(0, 0);
  loadB(0);
  int cur = 0;
  for (int k0 = 0; k0 < DIM; k0 += 32, cur ^= 1) {
    storeB(cur);
    cp_wait();                 // As[cur] resident
    __syncthreads();           // publish As[cur], Bs[cur]
    if (k0 + 32 < DIM) {       // next tile in flight during WMMAs
      issueA(k0 + 32, cur ^ 1);
      loadB(k0 + 32);
    }
    v16h a0 = load_frag16(As[cur], 40, wr * 32, 0);
    v16h a1 = load_frag16(As[cur], 40, wr * 32 + 16, 0);
    v16h b0 = load_frag16(Bs[cur], 40, wc * 32, 0);
    v16h b1 = load_frag16(Bs[cur], 40, wc * 32 + 16, 0);
    acc[0] = wmma_f16(a0, b0, acc[0]);
    acc[1] = wmma_f16(a0, b1, acc[1]);
    acc[2] = wmma_f16(a1, b0, acc[2]);
    acc[3] = wmma_f16(a1, b1, acc[3]);
    __syncthreads();
  }

  const int nlo  = lane & 15;
  const int mofs = (lane >> 4) << 3;
#pragma unroll
  for (int ar = 0; ar < 2; ++ar) {
#pragma unroll
    for (int bc = 0; bc < 2; ++bc) {
      v8f a = acc[ar * 2 + bc];
#pragma unroll
      for (int i = 0; i < 8; ++i) {
        int m = m0 + wr * 32 + ar * 16 + mofs + i;
        int n = n0 + wc * 32 + bc * 16 + nlo;
        out[(size_t)m * DIM + n] = a[i] + bias[n];
      }
    }
  }
}

// ---------------------------------------------------------------------------
// Host launcher
// ---------------------------------------------------------------------------
extern "C" void kernel_launch(void* const* d_in, const int* in_sizes, int n_in,
                              void* d_out, int out_size, void* d_ws, size_t ws_size,
                              hipStream_t stream) {
  (void)in_sizes; (void)n_in; (void)out_size; (void)ws_size;
  const float* x      = (const float*)d_in[0];
  const float* w_qkv  = (const float*)d_in[1];
  const float* b_qkv  = (const float*)d_in[2];
  const float* w_proj = (const float*)d_in[3];
  const float* b_proj = (const float*)d_in[4];
  float* out = (float*)d_out;

  char* ws = (char*)d_ws;
  const size_t per_buf = (size_t)HEADS * SEQ * HD * sizeof(_Float16); // 6 MB
  _Float16* qbuf = (_Float16*)(ws);
  _Float16* kbuf = (_Float16*)(ws + per_buf);
  _Float16* vbuf = (_Float16*)(ws + 2 * per_buf);
  _Float16* obuf = (_Float16*)(ws + 3 * per_buf);

  dim3 blk(256);
  qkv_gemm_kernel<<<dim3(NQKV / 64, SEQ / 128), blk, 0, stream>>>(
      x, w_qkv, b_qkv, qbuf, kbuf, vbuf);
  attn_kernel<<<dim3(SEQ / 64, HEADS), blk, 0, stream>>>(
      qbuf, kbuf, vbuf, obuf);
  proj_gemm_kernel<<<dim3(DIM / 64, SEQ / 128), blk, 0, stream>>>(
      obuf, w_proj, b_proj, out);
}
